// TreunoAttention_85710367359037
// MI455X (gfx1250) — compile-verified, matrix operand order checked by reading, not run on hardware
//
#include <hip/hip_runtime.h>
#include <hip/hip_bf16.h>

// ---------------------------------------------------------------------------
// CDNA5 (gfx1250) attention block: QKV proj + RoPE + causal GQA flash-attn +
// output proj, all matrix math on v_wmma_f32_16x16x32_bf16 (wave32).
// ---------------------------------------------------------------------------

typedef __bf16 bf16;
typedef __attribute__((ext_vector_type(16))) __bf16 v16bf;
typedef __attribute__((ext_vector_type(8)))  float  v8f;

#define BATCH 4
#define SEQ   2048
#define HID   768
#define NQH   12
#define NKVH  4
#define HD    64
#define ROWS  (BATCH * SEQ)   // 8192

union FragB { v16bf v; unsigned int w[8]; };

// Load a 16x32 bf16 WMMA operand fragment. For A: line = M row; for B:
// line = N column (B stored so K is contiguous per line). Per ISA layout,
// lane l -> line = l&15, half = l>>4; K u32-pair index for slot p:
//   ku = (p&3) + (p>=4 ? 8 : 0) + half*4
// (lowered by the compiler to two b128 loads per fragment)
__device__ __forceinline__ v16bf load_frag(const unsigned int* base, int strideU, int lane) {
  const int line = lane & 15;
  const int half = lane >> 4;
  FragB f;
  const unsigned int* row = base + (size_t)line * strideU;
#pragma unroll
  for (int p = 0; p < 8; ++p)
    f.w[p] = row[(p & 3) + ((p >> 2) << 3) + (half << 2)];
  return f.v;
}

__device__ __forceinline__ v8f wmma_bf16(v16bf a, v16bf b, v8f c) {
  return __builtin_amdgcn_wmma_f32_16x16x32_bf16(false, a, false, b, (short)0, c,
                                                 false, false);
}

// ------------------------------ conversions --------------------------------

__global__ void k_f32_to_bf16(const float* __restrict__ src, bf16* __restrict__ dst, int n) {
  int i = (blockIdx.x * blockDim.x + threadIdx.x) * 4;
  if (i + 3 < n) {
    float4 v = *(const float4*)(src + i);
    dst[i + 0] = (bf16)v.x;
    dst[i + 1] = (bf16)v.y;
    dst[i + 2] = (bf16)v.z;
    dst[i + 3] = (bf16)v.w;
  }
}

// W (K x N) f32 row-major  ->  Wt (N x K) bf16 row-major
__global__ void k_w_transpose_bf16(const float* __restrict__ W, bf16* __restrict__ Wt,
                                   int K, int N) {
  int idx = blockIdx.x * blockDim.x + threadIdx.x;
  if (idx < K * N) {
    int k = idx / N, n = idx % N;
    Wt[(size_t)n * K + k] = (bf16)W[idx];
  }
}

// ------------------------------ WMMA GEMM ----------------------------------
// C[M x N] = A[M x K] * Bt[N x K]^T ; one wave computes a 16x64 C strip
// (one A fragment reused across 4 B fragments per K-step). Software pipelined
// one K-step ahead so loads of step i+1 overlap the WMMAs of step i.
// mode 0/2: fp32 row-major out. mode 1: bf16 out to (b, kvh, s, d) V layout.
__global__ __launch_bounds__(256) void k_gemm(const bf16* __restrict__ A,
                                              const bf16* __restrict__ Bt,
                                              void* __restrict__ Cout,
                                              int M, int N, int K, int mode) {
  const int wid  = (blockIdx.x * blockDim.x + threadIdx.x) >> 5;
  const int lane = threadIdx.x & 31;
  const int tilesN4 = N >> 6;           // 64-wide strips
  const int tilesM  = M >> 4;
  if (wid >= tilesM * tilesN4) return;  // whole-wave uniform
  const int mt  = wid / tilesN4;
  const int nt4 = wid % tilesN4;

  const unsigned int* Abase = (const unsigned int*)(A + (size_t)mt * 16 * K);
  const unsigned int* B0 = (const unsigned int*)(Bt + (size_t)(nt4 * 64 +  0) * K);
  const unsigned int* B1 = (const unsigned int*)(Bt + (size_t)(nt4 * 64 + 16) * K);
  const unsigned int* B2 = (const unsigned int*)(Bt + (size_t)(nt4 * 64 + 32) * K);
  const unsigned int* B3 = (const unsigned int*)(Bt + (size_t)(nt4 * 64 + 48) * K);
  const int strideU = K >> 1;

  v8f acc0 = {0.f,0.f,0.f,0.f,0.f,0.f,0.f,0.f};
  v8f acc1 = acc0, acc2 = acc0, acc3 = acc0;

  // prologue: fragments for k0 = 0
  v16bf a  = load_frag(Abase, strideU, lane);
  v16bf b0 = load_frag(B0, strideU, lane);
  v16bf b1 = load_frag(B1, strideU, lane);
  v16bf b2 = load_frag(B2, strideU, lane);
  v16bf b3 = load_frag(B3, strideU, lane);

  for (int k0 = 0; k0 + 32 < K; k0 += 32) {
    const int u = (k0 + 32) >> 1;       // preload next K-step
    v16bf an  = load_frag(Abase + u, strideU, lane);
    v16bf bn0 = load_frag(B0 + u, strideU, lane);
    v16bf bn1 = load_frag(B1 + u, strideU, lane);
    v16bf bn2 = load_frag(B2 + u, strideU, lane);
    v16bf bn3 = load_frag(B3 + u, strideU, lane);
    acc0 = wmma_bf16(a, b0, acc0);
    acc1 = wmma_bf16(a, b1, acc1);
    acc2 = wmma_bf16(a, b2, acc2);
    acc3 = wmma_bf16(a, b3, acc3);
    a = an; b0 = bn0; b1 = bn1; b2 = bn2; b3 = bn3;
  }
  acc0 = wmma_bf16(a, b0, acc0);        // peeled last step
  acc1 = wmma_bf16(a, b1, acc1);
  acc2 = wmma_bf16(a, b2, acc2);
  acc3 = wmma_bf16(a, b3, acc3);

  const int half = lane >> 4, nl = lane & 15;
  v8f accs[4] = {acc0, acc1, acc2, acc3};
  if (mode != 1) {
    float* C = (float*)Cout;
#pragma unroll
    for (int j = 0; j < 4; ++j)
#pragma unroll
      for (int r = 0; r < 8; ++r) {
        int row = mt * 16 + r + 8 * half;
        C[(size_t)row * N + nt4 * 64 + j * 16 + nl] = accs[j][r];
      }
  } else {
    bf16* C = (bf16*)Cout;
#pragma unroll
    for (int j = 0; j < 4; ++j) {
      int ncol = nt4 * 64 + j * 16 + nl;
      int kvh = ncol >> 6, d = ncol & 63;
#pragma unroll
      for (int r = 0; r < 8; ++r) {
        int row = mt * 16 + r + 8 * half;
        int bb = row >> 11, s = row & 2047;
        C[(((size_t)(bb * NKVH + kvh) * SEQ) + s) * HD + d] = (bf16)accs[j][r];
      }
    }
  }
}

// -------------------------------- RoPE -------------------------------------
// src: (row, h, d) f32 with row = b*SEQ+s ; dst: (b, h, s, d) bf16.
__global__ void k_rope(const float* __restrict__ src, bf16* __restrict__ dst, int nheads) {
  int idx = blockIdx.x * blockDim.x + threadIdx.x;
  int total = ROWS * nheads * (HD / 2);
  if (idx >= total) return;
  int d = idx & 31;
  int h = (idx >> 5) % nheads;
  int row = idx / (32 * nheads);
  int b = row >> 11, s = row & 2047;
  const float* p = src + (size_t)row * nheads * HD + h * HD;
  float x1 = p[d], x2 = p[d + 32];
  float ang = (float)s * __powf(10000.0f, -(float)d * (1.0f / 32.0f));
  float sn, c;
  __sincosf(ang, &sn, &c);
  bf16* q = dst + (((size_t)(b * nheads + h) * SEQ) + s) * HD;
  q[d]      = (bf16)(x1 * c - x2 * sn);
  q[d + 32] = (bf16)(x2 * c + x1 * sn);
}

// ---------------------------- flash attention ------------------------------
// Grid (32, NQH, BATCH); 128 threads = 4 waves; wave owns 16 q rows,
// kv tiles of 64 positions (16 WMMAs per tile per wave).
__global__ __launch_bounds__(128) void k_flash(const bf16* __restrict__ Q,
                                               const bf16* __restrict__ Km,
                                               const bf16* __restrict__ V,
                                               bf16* __restrict__ O) {
  __shared__ __align__(16) unsigned short ldsK[64 * 64];   // (t, d)   8 KB
  __shared__ __align__(16) unsigned short ldsVt[64 * 64];  // (d, t)   8 KB
  __shared__ __align__(16) float scoreBuf[4][16][64];      //         16 KB
  __shared__ __align__(16) bf16  pBuf[4][16][64];          //          8 KB
  __shared__ float rowM[4][16], rowL[4][16], rowA[4][16];

  const int lane = threadIdx.x & 31;
  const int w    = threadIdx.x >> 5;     // 0..3
  const int qb   = blockIdx.x;           // 0..31 (64 q rows per block)
  const int qh   = blockIdx.y;
  const int b    = blockIdx.z;
  const int kvh  = qh / (NQH / NKVH);
  const int m0   = qb * 64 + w * 16;
  const int nl   = lane & 15, half = lane >> 4;

  // Q A-fragments (d 0..31 and 32..63), loop invariant
  const unsigned int* qbase =
      (const unsigned int*)(Q + (((size_t)(b * NQH + qh) * SEQ) + m0) * HD);
  v16bf qa0 = load_frag(qbase, 32, lane);
  v16bf qa1 = load_frag(qbase + 16, 32, lane);

  v8f acc0 = {0.f,0.f,0.f,0.f,0.f,0.f,0.f,0.f};
  v8f acc1 = acc0, acc2 = acc0, acc3 = acc0;

  rowM[w][nl] = -1e30f;
  rowL[w][nl] = 0.0f;

  const bf16* kb = Km + ((size_t)(b * NKVH + kvh) * SEQ) * HD;
  const bf16* vb = V  + ((size_t)(b * NKVH + kvh) * SEQ) * HD;
  const int kend = (qb + 1) * 64;

  for (int kt = 0; kt < kend; kt += 64) {
    __syncthreads();
    // stage K tile (64x64 bf16 = 8KB): 4 uint4 per thread
    {
      const uint4* gk = (const uint4*)(kb + (size_t)kt * HD);
#pragma unroll
      for (int i = 0; i < 4; ++i)
        ((uint4*)ldsK)[threadIdx.x + i * 128] = gk[threadIdx.x + i * 128];
    }
    // stage V tile transposed -> ldsVt[d][t]
    {
      int tr = threadIdx.x >> 1;         // 0..63  (kv position)
      int d0 = (threadIdx.x & 1) * 32;   // 0 or 32
#pragma unroll
      for (int c = 0; c < 4; ++c) {
        union { uint4 u; unsigned short h[8]; } uu;
        uu.u = *(const uint4*)(vb + ((size_t)(kt + tr)) * HD + d0 + c * 8);
#pragma unroll
        for (int j = 0; j < 8; ++j) ldsVt[(d0 + c * 8 + j) * 64 + tr] = uu.h[j];
      }
    }
    __syncthreads();

    // scores: S = Q * K^T, four 16x16 N-tiles, K-dim 64 = 2 WMMA each
#pragma unroll
    for (int ntile = 0; ntile < 4; ++ntile) {
      v8f sc = {0.f,0.f,0.f,0.f,0.f,0.f,0.f,0.f};
      v16bf kf0 = load_frag((const unsigned int*)ldsK + (ntile * 16) * 32, 32, lane);
      sc = wmma_bf16(qa0, kf0, sc);
      v16bf kf1 = load_frag((const unsigned int*)ldsK + (ntile * 16) * 32 + 16, 32, lane);
      sc = wmma_bf16(qa1, kf1, sc);
#pragma unroll
      for (int r = 0; r < 8; ++r) {
        int mrow = r + 8 * half;
        int srow = m0 + mrow;
        int tcol = kt + ntile * 16 + nl;
        float valf = sc[r] * 0.125f;                 // 1/sqrt(64)
        if (tcol > srow) valf = -1e30f;              // causal mask
        scoreBuf[w][mrow][ntile * 16 + nl] = valf;
      }
    }
    asm volatile("s_wait_dscnt 0" ::: "memory");     // wave-local LDS ordering

    // online softmax: lane handles row nl, columns [half*32, half*32+32);
    // halves merged with one cross-lane xor-shuffle.
    {
      float mold = rowM[w][nl];
      const int c0 = half * 32;
      float pmax = -1e30f;
#pragma unroll
      for (int j = 0; j < 32; ++j) pmax = fmaxf(pmax, scoreBuf[w][nl][c0 + j]);
      float tmax = fmaxf(pmax, __shfl_xor(pmax, 16));
      float mnew = fmaxf(mold, tmax);
      float alpha = __expf(mold - mnew);
      float psum = 0.0f;
#pragma unroll
      for (int j = 0; j < 32; ++j) {
        float pe = __expf(scoreBuf[w][nl][c0 + j] - mnew);
        psum += pe;
        pBuf[w][nl][c0 + j] = (bf16)pe;
      }
      float lsum = psum + __shfl_xor(psum, 16);
      rowM[w][nl] = mnew;                    // both halves write same values
      rowL[w][nl] = rowL[w][nl] * alpha + lsum;
      rowA[w][nl] = alpha;
    }
    asm volatile("s_wait_dscnt 0" ::: "memory");

    // rescale accumulators, then O += P * V (K-dim 64 = 2 chained WMMA x 4 N)
    float af[8];
#pragma unroll
    for (int r = 0; r < 8; ++r) af[r] = rowA[w][r + 8 * half];
#pragma unroll
    for (int r = 0; r < 8; ++r) {
      acc0[r] *= af[r]; acc1[r] *= af[r]; acc2[r] *= af[r]; acc3[r] *= af[r];
    }
    v16bf pa0 = load_frag((const unsigned int*)&pBuf[w][0][0], 32, lane);
    v16bf pa1 = load_frag((const unsigned int*)&pBuf[w][0][0] + 16, 32, lane);

    v16bf v00 = load_frag((const unsigned int*)ldsVt + 0 * 512, 32, lane);
    acc0 = wmma_bf16(pa0, v00, acc0);
    v16bf v01 = load_frag((const unsigned int*)ldsVt + 0 * 512 + 16, 32, lane);
    acc0 = wmma_bf16(pa1, v01, acc0);
    v16bf v10 = load_frag((const unsigned int*)ldsVt + 1 * 512, 32, lane);
    acc1 = wmma_bf16(pa0, v10, acc1);
    v16bf v11 = load_frag((const unsigned int*)ldsVt + 1 * 512 + 16, 32, lane);
    acc1 = wmma_bf16(pa1, v11, acc1);
    v16bf v20 = load_frag((const unsigned int*)ldsVt + 2 * 512, 32, lane);
    acc2 = wmma_bf16(pa0, v20, acc2);
    v16bf v21 = load_frag((const unsigned int*)ldsVt + 2 * 512 + 16, 32, lane);
    acc2 = wmma_bf16(pa1, v21, acc2);
    v16bf v30 = load_frag((const unsigned int*)ldsVt + 3 * 512, 32, lane);
    acc3 = wmma_bf16(pa0, v30, acc3);
    v16bf v31 = load_frag((const unsigned int*)ldsVt + 3 * 512 + 16, 32, lane);
    acc3 = wmma_bf16(pa1, v31, acc3);
  }

  asm volatile("s_wait_dscnt 0" ::: "memory");
  float invr[8];
#pragma unroll
  for (int r = 0; r < 8; ++r) invr[r] = 1.0f / rowL[w][r + 8 * half];

  // write attn output as (b*SEQ+s, qh*64+d) bf16 rows for the final GEMM
  bf16* obase = O + ((size_t)(b * SEQ) + m0) * (NQH * HD) + qh * HD;
#pragma unroll
  for (int r = 0; r < 8; ++r) {
    bf16* orow = obase + (size_t)(r + 8 * half) * (NQH * HD);
    orow[ 0 + nl] = (bf16)(acc0[r] * invr[r]);
    orow[16 + nl] = (bf16)(acc1[r] * invr[r]);
    orow[32 + nl] = (bf16)(acc2[r] * invr[r]);
    orow[48 + nl] = (bf16)(acc3[r] * invr[r]);
  }
}

// ------------------------------- launcher ----------------------------------

extern "C" void kernel_launch(void* const* d_in, const int* in_sizes, int n_in,
                              void* d_out, int out_size, void* d_ws, size_t ws_size,
                              hipStream_t stream) {
  (void)in_sizes; (void)n_in; (void)out_size; (void)ws_size;
  const float* hs = (const float*)d_in[0];
  const float* Wq = (const float*)d_in[1];
  const float* Wk = (const float*)d_in[2];
  const float* Wv = (const float*)d_in[3];
  const float* Wo = (const float*)d_in[4];

  char* ws = (char*)d_ws;
  size_t off = 0;
  auto take = [&](size_t bytes) -> char* {
    off = (off + 255) & ~(size_t)255;
    char* p = ws + off;
    off += bytes;
    return p;
  };

  bf16*  Xbf  = (bf16*) take((size_t)ROWS * HID * 2);
  bf16*  Wqt  = (bf16*) take((size_t)768 * 768 * 2);
  bf16*  Wkt  = (bf16*) take((size_t)256 * 768 * 2);
  bf16*  Wvt  = (bf16*) take((size_t)256 * 768 * 2);
  bf16*  Wot  = (bf16*) take((size_t)768 * 768 * 2);
  float* qtmp = (float*)take((size_t)ROWS * 768 * 4);
  float* ktmp = (float*)take((size_t)ROWS * 256 * 4);
  bf16*  Qbf  = (bf16*) take((size_t)ROWS * 768 * 2);
  bf16*  Kbf  = (bf16*) take((size_t)ROWS * 256 * 2);
  bf16*  Vbf  = (bf16*) take((size_t)ROWS * 256 * 2);
  bf16*  Abf  = (bf16*) take((size_t)ROWS * 768 * 2);

  // conversions
  k_f32_to_bf16<<<(ROWS * HID / 4 + 255) / 256, 256, 0, stream>>>(hs, Xbf, ROWS * HID);
  k_w_transpose_bf16<<<(768 * 768 + 255) / 256, 256, 0, stream>>>(Wq, Wqt, 768, 768);
  k_w_transpose_bf16<<<(768 * 256 + 255) / 256, 256, 0, stream>>>(Wk, Wkt, 768, 256);
  k_w_transpose_bf16<<<(768 * 256 + 255) / 256, 256, 0, stream>>>(Wv, Wvt, 768, 256);
  k_w_transpose_bf16<<<(768 * 768 + 255) / 256, 256, 0, stream>>>(Wo, Wot, 768, 768);

  // projections (WMMA), 16x64 C strip per wave
  const int tq = (ROWS / 16) * (768 / 64);
  const int tk = (ROWS / 16) * (256 / 64);
  k_gemm<<<(tq + 7) / 8, 256, 0, stream>>>(Xbf, Wqt, (void*)qtmp, ROWS, 768, 768, 0);
  k_gemm<<<(tk + 7) / 8, 256, 0, stream>>>(Xbf, Wkt, (void*)ktmp, ROWS, 256, 768, 0);
  k_gemm<<<(tk + 7) / 8, 256, 0, stream>>>(Xbf, Wvt, (void*)Vbf,  ROWS, 256, 768, 1);

  // RoPE -> bf16 per-head layouts
  k_rope<<<(ROWS * NQH  * 32 + 255) / 256, 256, 0, stream>>>(qtmp, Qbf, NQH);
  k_rope<<<(ROWS * NKVH * 32 + 255) / 256, 256, 0, stream>>>(ktmp, Kbf, NKVH);

  // causal GQA flash attention (WMMA), 64 q rows / 4 waves per workgroup
  dim3 g(SEQ / 64, NQH, BATCH);
  k_flash<<<g, 128, 0, stream>>>(Qbf, Kbf, Vbf, Abf);

  // output projection (WMMA) -> fp32 d_out
  k_gemm<<<(tq + 7) / 8, 256, 0, stream>>>(Abf, Wot, d_out, ROWS, 768, 768, 2);
}
